// GAT_5351529251344
// MI455X (gfx1250) — matrix operand order repaired and checked by reading, not compiled, not run on hardware
//
#include <hip/hip_runtime.h>
#include <hip/hip_bf16.h>
#include <stdint.h>

// ---------------------------------------------------------------------------
// Types for CDNA5 WMMA
// ---------------------------------------------------------------------------
typedef __attribute__((ext_vector_type(16))) __bf16 v16bf;
typedef __attribute__((ext_vector_type(8)))  __bf16 v8bf;
typedef __attribute__((ext_vector_type(8)))  float  v8f;

#define N_NODES 50000
#define N_EDGES 500000

static __device__ __forceinline__ v16bf cat8(v8bf lo, v8bf hi) {
    return __builtin_shufflevector(lo, hi, 0, 1, 2, 3, 4, 5, 6, 7,
                                           8, 9, 10, 11, 12, 13, 14, 15);
}

// ---------------------------------------------------------------------------
// Tiled GEMM:  C[M,Nout] = A[M,K](f32) * B[K,Nw](f32) (+bias)
// via v_wmma_f32_16x16x32_bf16.  K must be a multiple of 32.
// Block tile 128x64, 256 threads = 8 waves (wave32):
//   4 waves along M (32 rows each) x 2 waves along N (32 cols each)
//   -> 4 accumulators (2 A-frags x 2 B-frags) = 4 WMMA per k-step per wave.
// LDS: A row-major [128][40], B transposed [64][40] (pad keeps 16B alignment,
// breaks bank conflicts). Fragments are contiguous b128 LDS loads.
// ---------------------------------------------------------------------------
#define BM 128
#define BN 64
#define BK 32
#define LPAD 8   // row pad in halves; stride 40 halves = 80 B (multiple of 16)

__global__ __launch_bounds__(256)
void gemm_wmma(const float* __restrict__ A, const float* __restrict__ B,
               const float* __restrict__ bias, float* __restrict__ C,
               int M, int K, int Nw, int Nout, int ldc)
{
    __shared__ __bf16 sA[BM][BK + LPAD];    // 128x40 bf16 = 10 KB
    __shared__ __bf16 sBt[BN][BK + LPAD];   // 64x40  bf16 = 5 KB (B transposed)

    const int tid  = threadIdx.x;
    const int wave = tid >> 5;
    const int lane = tid & 31;
    const int m0 = blockIdx.x * BM;
    const int n0 = blockIdx.y * BN;

    const int wm = (wave & 3) * 32;     // wave row offset in block
    const int wn = (wave >> 2) * 32;    // wave col offset in block

    const int row16 = lane & 15;
    const int khi   = (lane >> 4) & 1;

    // B staging mapping: tid -> (column c, k-chunk) ; coalesced over c
    const int bc    = tid & 63;
    const int bkc   = tid >> 6;         // 4 chunks of 8 k-values
    const int gn_b  = n0 + bc;
    const float bscale = (gn_b < Nw) ? 1.0f : 0.0f;
    const int gn_bc = (gn_b < Nw) ? gn_b : (Nw - 1);

    v8f acc00 = {}, acc01 = {}, acc10 = {}, acc11 = {};

    for (int k0 = 0; k0 < K; k0 += BK) {
        // ---- stage A tile (128 x 32): 512 (row,seg8) units, 2 per thread ----
        #pragma unroll
        for (int u = 0; u < 2; ++u) {
            int i   = tid + u * 256;
            int r   = i >> 2;
            int seg = i & 3;
            int gm  = m0 + r;
            float scale = (gm < M) ? 1.0f : 0.0f;
            int gmc = (gm < M) ? gm : (M - 1);
            const float4* p =
                (const float4*)(A + (int64_t)gmc * K + k0 + seg * 8);
            float4 x = p[0];
            float4 y = p[1];
            v8bf pk;
            pk[0] = (__bf16)(x.x * scale); pk[1] = (__bf16)(x.y * scale);
            pk[2] = (__bf16)(x.z * scale); pk[3] = (__bf16)(x.w * scale);
            pk[4] = (__bf16)(y.x * scale); pk[5] = (__bf16)(y.y * scale);
            pk[6] = (__bf16)(y.z * scale); pk[7] = (__bf16)(y.w * scale);
            *(v8bf*)&sA[r][seg * 8] = pk;
        }
        // ---- stage B tile (32 x 64) transposed: 1 unit per thread ----
        {
            v8bf pk;
            const float* bp = B + (int64_t)(k0 + bkc * 8) * Nw + gn_bc;
            #pragma unroll
            for (int j = 0; j < 8; ++j)
                pk[j] = (__bf16)(bp[(int64_t)j * Nw] * bscale);
            *(v8bf*)&sBt[bc][bkc * 8] = pk;
        }
        // prefetch next A tile (gfx1250 global_prefetch path)
        if (k0 + BK < K) {
            int gm = m0 + (tid >> 1);
            if (gm < M) __builtin_prefetch(A + (int64_t)gm * K + (k0 + BK), 0, 1);
        }
        __syncthreads();

        // ---- fragments: contiguous b128 LDS loads ----
        // A 16x32 frag: halves 0..7 -> K = khi*8+0..7 ; 8..15 -> 16+khi*8+0..7
        v16bf a0 = cat8(*(const v8bf*)&sA[wm + row16][khi * 8],
                        *(const v8bf*)&sA[wm + row16][16 + khi * 8]);
        v16bf a1 = cat8(*(const v8bf*)&sA[wm + 16 + row16][khi * 8],
                        *(const v8bf*)&sA[wm + 16 + row16][16 + khi * 8]);
        // B 32x16 frag: col = lane&15 ; halves 0..7 -> K = khi*8+0..7 ; 8..15 -> 16+khi*8+0..7
        v16bf b0 = cat8(*(const v8bf*)&sBt[wn + row16][khi * 8],
                        *(const v8bf*)&sBt[wn + row16][16 + khi * 8]);
        v16bf b1 = cat8(*(const v8bf*)&sBt[wn + 16 + row16][khi * 8],
                        *(const v8bf*)&sBt[wn + 16 + row16][16 + khi * 8]);

        acc00 = __builtin_amdgcn_wmma_f32_16x16x32_bf16(
                    false, a0, false, b0, (short)0, acc00, false, false);
        acc01 = __builtin_amdgcn_wmma_f32_16x16x32_bf16(
                    false, a0, false, b1, (short)0, acc01, false, false);
        acc10 = __builtin_amdgcn_wmma_f32_16x16x32_bf16(
                    false, a1, false, b0, (short)0, acc10, false, false);
        acc11 = __builtin_amdgcn_wmma_f32_16x16x32_bf16(
                    false, a1, false, b1, (short)0, acc11, false, false);
        __syncthreads();
    }

    // ---- store: C/D layout: VGPR r, lanes 0-15 -> M=r, 16-31 -> M=r+8 ------
    const int col0 = n0 + wn + row16;
    const int col1 = col0 + 16;
    const float bias0 = (bias && col0 < Nout) ? bias[col0] : 0.0f;
    const float bias1 = (bias && col1 < Nout) ? bias[col1] : 0.0f;
    #pragma unroll
    for (int mi = 0; mi < 2; ++mi) {
        const v8f& accL = mi ? acc10 : acc00;
        const v8f& accR = mi ? acc11 : acc01;
        int mbase = m0 + wm + mi * 16 + (khi << 3);
        #pragma unroll
        for (int r = 0; r < 8; ++r) {
            int gm = mbase + r;
            if (gm < M) {
                if (col0 < Nout) C[(int64_t)gm * ldc + col0] = accL[r] + bias0;
                if (col1 < Nout) C[(int64_t)gm * ldc + col1] = accR[r] + bias1;
            }
        }
    }
}

// ---------------------------------------------------------------------------
// el/er:  out[n,h] = sum_{d<Ddot} feat[n*ldf + h*hs + d] * attn[h*Ddot + d]
// ---------------------------------------------------------------------------
__global__ __launch_bounds__(256)
void attn_dot(const float* __restrict__ feat, const float* __restrict__ attn,
              float* __restrict__ out, int Nn, int H, int Ddot, int ldf, int hs)
{
    int idx = blockIdx.x * blockDim.x + threadIdx.x;
    if (idx >= Nn * H) return;
    int n = idx / H, h = idx % H;
    const float* f = feat + (int64_t)n * ldf + h * hs;
    const float* a = attn + h * Ddot;
    float s = 0.f;
    #pragma unroll 4
    for (int d = 0; d < Ddot; ++d) s += f[d] * a[d];
    out[idx] = s;
}

// ---------------------------------------------------------------------------
// order-preserving float <-> uint mapping for atomicMax-based segment max
// ---------------------------------------------------------------------------
__device__ __forceinline__ unsigned fenc(float f) {
    unsigned u = __float_as_uint(f);
    return (u & 0x80000000u) ? ~u : (u | 0x80000000u);
}
__device__ __forceinline__ float fdec(unsigned u) {
    return (u & 0x80000000u) ? __uint_as_float(u & 0x7FFFFFFFu)
                             : __uint_as_float(~u);
}

// pass 1: e = leaky_relu(el[src]+er[dst]); store raw e; segment max over dst
__global__ __launch_bounds__(256)
void edge_logit_max(const float* __restrict__ el, const float* __restrict__ er,
                    const int* __restrict__ src, const int* __restrict__ dst,
                    float* __restrict__ eraw, unsigned* __restrict__ emax,
                    int E, int H)
{
    int idx = blockIdx.x * blockDim.x + threadIdx.x;
    if (idx >= E * H) return;
    int e = idx / H, h = idx % H;
    int s = src[e], d = dst[e];
    float v = el[s * H + h] + er[d * H + h];
    v = (v > 0.f) ? v : 0.2f * v;           // leaky_relu, slope 0.2
    eraw[idx] = v;
    atomicMax(&emax[d * H + h], fenc(v));
}

// pass 2: ee = exp(e - emax[dst]); denom[dst] += ee
__global__ __launch_bounds__(256)
void edge_exp_sum(float* __restrict__ eraw, const unsigned* __restrict__ emax,
                  const int* __restrict__ dst, float* __restrict__ denom,
                  int E, int H)
{
    int idx = blockIdx.x * blockDim.x + threadIdx.x;
    if (idx >= E * H) return;
    int e = idx / H, h = idx % H;
    int d = dst[e];
    float m  = fdec(emax[d * H + h]);
    float ee = __expf(eraw[idx] - m);
    eraw[idx] = ee;
    atomicAdd(&denom[d * H + h], ee);
}

// pass 3: out[dst,h,:] += feat[src,h,:] * (ee / max(denom,1e-9)), float4 granules
__global__ __launch_bounds__(256)
void edge_aggregate(const float* __restrict__ feat, const float* __restrict__ ee,
                    const float* __restrict__ denom,
                    const int* __restrict__ src, const int* __restrict__ dst,
                    float* __restrict__ out, int E, int H, int D, int ldf)
{
    int64_t idx = (int64_t)blockIdx.x * blockDim.x + threadIdx.x;
    int d4c = D >> 2;
    int64_t total = (int64_t)E * H * d4c;
    if (idx >= total) return;
    int d4 = (int)(idx % d4c);
    int64_t eh = idx / d4c;
    int h = (int)(eh % H);
    int e = (int)(eh / H);
    int s = src[e], t = dst[e];
    float den   = denom[t * H + h];
    float alpha = ee[e * H + h] / fmaxf(den, 1e-9f);
    const float4 f = *(const float4*)(feat + (int64_t)s * ldf + h * D + d4 * 4);
    float* o = out + (int64_t)t * ldf + h * D + d4 * 4;
    atomicAdd(o + 0, f.x * alpha);
    atomicAdd(o + 1, f.y * alpha);
    atomicAdd(o + 2, f.z * alpha);
    atomicAdd(o + 3, f.w * alpha);
}

// bias + ELU (in place)
__global__ __launch_bounds__(256)
void bias_elu(float* __restrict__ x, const float* __restrict__ bias,
              int64_t n, int ld)
{
    int64_t idx = (int64_t)blockIdx.x * blockDim.x + threadIdx.x;
    if (idx >= n) return;
    int c = (int)(idx % ld);
    float v = x[idx] + bias[c];
    x[idx] = (v > 0.f) ? v : (__expf(v) - 1.f);
}

// final: logits[n,c] = agg[n*16+c] + b2[c]   (mean over 1 head == identity)
__global__ __launch_bounds__(256)
void final_out(const float* __restrict__ agg, const float* __restrict__ b2,
               float* __restrict__ out, int Nn)
{
    int idx = blockIdx.x * blockDim.x + threadIdx.x;
    if (idx >= Nn * 10) return;
    int n = idx / 10, c = idx % 10;
    out[idx] = agg[n * 16 + c] + b2[c];
}

// ---------------------------------------------------------------------------
// Host side
// ---------------------------------------------------------------------------
static inline int cdiv(int64_t a, int64_t b) { return (int)((a + b - 1) / b); }

static void run_gat_layer(const float* h, int K, const float* W,
                          const float* al, const float* ar,
                          int H, int D, int Ddot, int Nw,
                          const int* src, const int* dst,
                          float* featb, float* el, float* er,
                          unsigned* emax, float* den, float* ee,
                          float* outb, hipStream_t stream)
{
    const int ldf = H * D;
    gemm_wmma<<<dim3(cdiv(N_NODES, BM), cdiv(ldf, BN)), 256, 0, stream>>>(
        h, W, nullptr, featb, N_NODES, K, Nw, ldf, ldf);

    const int nh = N_NODES * H;
    attn_dot<<<cdiv(nh, 256), 256, 0, stream>>>(featb, al, el, N_NODES, H, Ddot, ldf, D);
    attn_dot<<<cdiv(nh, 256), 256, 0, stream>>>(featb, ar, er, N_NODES, H, Ddot, ldf, D);

    hipMemsetAsync(emax, 0, (size_t)nh * sizeof(unsigned), stream);
    hipMemsetAsync(den,  0, (size_t)nh * sizeof(float), stream);

    const int eh = N_EDGES * H;
    edge_logit_max<<<cdiv(eh, 256), 256, 0, stream>>>(el, er, src, dst, ee, emax, N_EDGES, H);
    edge_exp_sum<<<cdiv(eh, 256), 256, 0, stream>>>(ee, emax, dst, den, N_EDGES, H);

    hipMemsetAsync(outb, 0, (size_t)N_NODES * ldf * sizeof(float), stream);
    int64_t total = (int64_t)N_EDGES * H * (D / 4);
    edge_aggregate<<<cdiv(total, 256), 256, 0, stream>>>(
        featb, ee, den, src, dst, outb, N_EDGES, H, D, ldf);
}

extern "C" void kernel_launch(void* const* d_in, const int* in_sizes, int n_in,
                              void* d_out, int out_size, void* d_ws, size_t ws_size,
                              hipStream_t stream)
{
    const float* feat0 = (const float*)d_in[0];   // [30000,256]
    const float* feat1 = (const float*)d_in[1];   // [20000,128]
    const float* fc0_w = (const float*)d_in[2];   // [256,64]
    const float* fc0_b = (const float*)d_in[3];   // [64]
    const float* fc1_w = (const float*)d_in[4];   // [128,64]
    const float* fc1_b = (const float*)d_in[5];   // [64]
    const float* W0    = (const float*)d_in[6];   // [64,512]
    const float* al0   = (const float*)d_in[7];   // [8,64]
    const float* ar0   = (const float*)d_in[8];
    const float* b0    = (const float*)d_in[9];   // [512]
    const float* W1    = (const float*)d_in[10];  // [512,512]
    const float* al1   = (const float*)d_in[11];
    const float* ar1   = (const float*)d_in[12];
    const float* b1    = (const float*)d_in[13];
    const float* W2    = (const float*)d_in[14];  // [512,10]
    const float* al2   = (const float*)d_in[15];  // [1,10]
    const float* ar2   = (const float*)d_in[16];
    const float* b2    = (const float*)d_in[17];  // [10]
    const int*   edge  = (const int*)d_in[18];    // [2, 500000]
    const int* src = edge;
    const int* dst = edge + N_EDGES;

    // workspace carve-up (bytes)
    const size_t SZ_BIG = (size_t)N_NODES * 512 * sizeof(float);   // 102.4 MB
    const size_t SZ_NH  = (size_t)N_NODES * 8 * sizeof(float);     // 1.6 MB
    char* w = (char*)d_ws;
    float*    bufA = (float*)(w);                       // h / aggregation target
    float*    bufB = (float*)(w + SZ_BIG);              // post-GEMM features
    float*    el   = (float*)(w + 2 * SZ_BIG);
    float*    er   = (float*)(w + 2 * SZ_BIG + SZ_NH);
    unsigned* emax = (unsigned*)(w + 2 * SZ_BIG + 2 * SZ_NH);
    float*    den  = (float*)(w + 2 * SZ_BIG + 3 * SZ_NH);
    float*    ee   = (float*)(w + 2 * SZ_BIG + 4 * SZ_NH);
    (void)ws_size; (void)in_sizes; (void)n_in; (void)out_size;

    // --- per-type projections -> h [50000,64] in bufA ---
    gemm_wmma<<<dim3(cdiv(30000, BM), 1), 256, 0, stream>>>(
        feat0, fc0_w, fc0_b, bufA, 30000, 256, 64, 64, 64);
    gemm_wmma<<<dim3(cdiv(20000, BM), 1), 256, 0, stream>>>(
        feat1, fc1_w, fc1_b, bufA + (int64_t)30000 * 64, 20000, 128, 64, 64, 64);

    // --- GAT layer 0: h[.,64] -> [.,8,64], ELU ---
    run_gat_layer(bufA, 64, W0, al0, ar0, /*H=*/8, /*D=*/64, /*Ddot=*/64, /*Nw=*/512,
                  src, dst, bufB, el, er, emax, den, ee, bufA, stream);
    bias_elu<<<cdiv((int64_t)N_NODES * 512, 256), 256, 0, stream>>>(
        bufA, b0, (int64_t)N_NODES * 512, 512);

    // --- GAT layer 1: [.,512] -> [.,8,64], ELU ---
    run_gat_layer(bufA, 512, W1, al1, ar1, 8, 64, 64, 512,
                  src, dst, bufB, el, er, emax, den, ee, bufA, stream);
    bias_elu<<<cdiv((int64_t)N_NODES * 512, 256), 256, 0, stream>>>(
        bufA, b1, (int64_t)N_NODES * 512, 512);

    // --- GAT layer 2: [.,512] -> [.,1,10] (padded to 16), no activation ---
    run_gat_layer(bufA, 512, W2, al2, ar2, /*H=*/1, /*D=*/16, /*Ddot=*/10, /*Nw=*/10,
                  src, dst, bufB, el, er, emax, den, ee, bufA, stream);

    final_out<<<cdiv(N_NODES * 10, 256), 256, 0, stream>>>(
        bufA, b2, (float*)d_out, N_NODES);
}
